// EGNN_dynamics_QM9_11330123727058
// MI455X (gfx1250) — matrix operand order repaired and checked by reading, not compiled, not run on hardware
//
#include <hip/hip_runtime.h>
#include <math.h>

// ---------------- types ----------------
typedef __attribute__((ext_vector_type(16))) _Float16 v16h;
typedef __attribute__((ext_vector_type(8)))  _Float16 v8h;
typedef __attribute__((ext_vector_type(8)))  float    v8f;
typedef __attribute__((ext_vector_type(4)))  unsigned int v4u;
typedef __attribute__((ext_vector_type(8)))  int      v8i;
typedef __attribute__((ext_vector_type(4)))  int      v4i;

// Problem constants
#define BSZ 32
#define NN  64
#define HD  8
#define HID 256
#define NL  4
#define NDIM 3
#define BN  (BSZ*NN)        // 2048 nodes
#define NE  (BSZ*NN*NN)     // 131072 edges

#define HAS_TDM __has_builtin(__builtin_amdgcn_tensor_load_to_lds)

__device__ __forceinline__ float silu_f(float x) {
    return x / (1.f + __expf(-x));
}

// ---------------- WMMA GEMM ----------------
// C(M x 256) = A(M x K, f16 row-major lda) * W(K x 256)  with W given PRE-TRANSPOSED
// as Bt[n][k] f16 (256 rows, row stride K).
// block = 128 threads (4 waves). Wave w: rows [row0+16w,+16) x 64 cols. grid = (M/64, 4).
// Per K-step the 64(n) x 32(k) f16 tile is staged into LDS by the Tensor Data Mover
// (TDM pads each 64B row by 16B -> 80B row stride == Bs[64][40] halfs).
// EPI: 0 plain f32 out; 1 silu(+bias)->f16; 2 silu(+bias)*rowmask->f16;
//      3 (resid+acc+bias)*rowmask -> f32 AND f16 out
template<int EPI>
__global__ __launch_bounds__(128)
void gemm_wmma(const _Float16* __restrict__ A, int lda,
               const _Float16* __restrict__ Bt, int K,
               const float* __restrict__ bias,
               const float* __restrict__ rowmask,
               const float* __restrict__ resid,
               float* __restrict__ outf,
               _Float16* __restrict__ outh, int ldo)
{
    __shared__ __align__(16) _Float16 Bs[64][40];   // [n][k], 80B row stride

    const int lane = threadIdx.x & 31;
    const int wid  = threadIdx.x >> 5;
    const int lrow = lane & 15;
    const int hi   = lane >> 4;
    const int row0 = blockIdx.x * 64;
    const int col0 = blockIdx.y * 64;
    const int wrow = row0 + wid * 16;

    v8f acc[4] = {};

    const _Float16* wsrc = Bt + (size_t)col0 * K;           // 64 rows of the weight panel
    unsigned lds_base = (unsigned)(size_t)(&Bs[0][0]);      // LDS offset (low 32 bits)

    for (int k0 = 0; k0 < K; k0 += 32) {
        __syncthreads();                                    // previous tile fully consumed
#if HAS_TDM
        if (wid == 0) {
            unsigned long long ga = (unsigned long long)(size_t)(wsrc + k0);
            v4u g0;
            g0[0] = 1u;                                     // count=1 (valid descriptor)
            g0[1] = lds_base;                               // lds_addr
            g0[2] = (unsigned)(ga & 0xffffffffu);           // global_addr[31:0]
            g0[3] = (unsigned)((ga >> 32) & 0x01ffffffu)    // global_addr[56:32]
                  | (2u << 30);                             // type=2 (image)
            v8i g1;
            g1[0] = (1 << 16)                               // data_size = 2 bytes
                  | (1 << 20)                               // pad_enable
                  | (3 << 22)                               // pad_interval: 16 DWORDs (64B)
                  | (3 << 25);                              // pad_amount: 4 DWORDs (16B)
            g1[1] = (int)(32u << 16);                       // tensor_dim0 = 32 (low16 @bit48)
            g1[2] = (int)(64u << 16);                       // tensor_dim1 = 64 (low16 @bit80)
            g1[3] = (int)(32u << 16);                       // tile_dim0 = 32 (@bit112)
            g1[4] = 64;                                     // tile_dim1 = 64, tile_dim2 = 0
            g1[5] = K;                                      // tensor_dim0_stride[31:0] = K
            g1[6] = 0;                                      // strides hi = 0
            g1[7] = 0;
            v4i gz4 = {};
            v8i gz8 = {};
            __builtin_amdgcn_tensor_load_to_lds(g0, g1, gz4, gz4, gz8, 0);
            __builtin_amdgcn_s_wait_tensorcnt(0);
        }
#else
        {
#pragma unroll
            for (int u = 0; u < 2; ++u) {
                int unit = threadIdx.x * 2 + u;             // 256 chunks of 8 halfs
                int n = unit >> 2, c = unit & 3;
                *reinterpret_cast<v8h*>(&Bs[n][c * 8]) =
                    *reinterpret_cast<const v8h*>(wsrc + (size_t)n * K + k0 + c * 8);
            }
        }
#endif
        __syncthreads();                                    // tile visible to all waves

        // A fragment: lane<16 -> K {k0..k0+7, k0+16..k0+23}; lane>=16 -> +8
        const _Float16* ap = A + (size_t)(wrow + lrow) * lda + k0 + hi * 8;
        __builtin_prefetch(ap + 32, 0, 0);                  // next K chunk
        v8h a_lo = *reinterpret_cast<const v8h*>(ap);
        v8h a_hi = *reinterpret_cast<const v8h*>(ap + 16);
        v16h a;
#pragma unroll
        for (int i = 0; i < 8; ++i) { a[i] = a_lo[i]; a[i + 8] = a_hi[i]; }

#pragma unroll
        for (int t = 0; t < 4; ++t) {
            // B fragment: column n = 16t + lrow; lane<16 -> K {0..7, 16..23}, lane>=16 -> +8
            const _Float16* bp = &Bs[t * 16 + lrow][hi * 8];
            v8h b_lo = *reinterpret_cast<const v8h*>(bp);
            v8h b_hi = *reinterpret_cast<const v8h*>(bp + 16);
            v16h b;
#pragma unroll
            for (int i = 0; i < 8; ++i) { b[i] = b_lo[i]; b[i + 8] = b_hi[i]; }
            acc[t] = __builtin_amdgcn_wmma_f32_16x16x32_f16(
                false, a, false, b, (short)0, acc[t], false, false);
        }
    }

    // Epilogue. C/D layout: VGPR r -> M = r (lanes 0-15) / 8+r (lanes 16-31); N = lane&15
#pragma unroll
    for (int t = 0; t < 4; ++t) {
#pragma unroll
        for (int r = 0; r < 8; ++r) {
            int grow = wrow + hi * 8 + r;
            int gcol = col0 + t * 16 + lrow;
            float v = acc[t][r];
            if constexpr (EPI == 0) {
                outf[(size_t)grow * 256 + gcol] = v;
            } else if constexpr (EPI == 1) {
                v = silu_f(v + bias[gcol]);
                outh[(size_t)grow * ldo + gcol] = (_Float16)v;
            } else if constexpr (EPI == 2) {
                v = silu_f(v + bias[gcol]) * rowmask[grow];
                outh[(size_t)grow * ldo + gcol] = (_Float16)v;
            } else {
                v = (resid[(size_t)grow * 256 + gcol] + v + bias[gcol]) * rowmask[grow];
                outf[(size_t)grow * 256 + gcol] = v;
                outh[(size_t)grow * ldo + gcol] = (_Float16)v;
            }
        }
    }
}

// ---------------- weight convert + transpose ----------------
// W (K x 256 f32, row-major) -> Wt (256 x K f16, row-major)
__global__ void wconv_kernel(const float* __restrict__ W, _Float16* __restrict__ Wt, int K)
{
    int k = blockIdx.x, n = threadIdx.x;    // K blocks x 256 threads
    Wt[(size_t)n * K + k] = (_Float16)W[(size_t)k * 256 + n];
}

// ---------------- small kernels ----------------

__global__ void preprocess_embed(const float* __restrict__ xh, const float* __restrict__ nm,
                                 const float* __restrict__ tsc,
                                 const float* __restrict__ W_emb, const float* __restrict__ b_emb,
                                 float* __restrict__ h, _Float16* __restrict__ hc,
                                 float* __restrict__ x0, float* __restrict__ xcur)
{
    int node = blockIdx.x, n = threadIdx.x;   // 2048 x 256
    float mask = nm[node];
    float s = b_emb[n];
#pragma unroll
    for (int k = 0; k < HD; ++k)
        s += xh[node * (NDIM + HD) + NDIM + k] * mask * W_emb[k * 256 + n];
    s += tsc[0] * W_emb[HD * 256 + n];
    h[(size_t)node * 256 + n] = s;
    hc[(size_t)node * 512 + n] = (_Float16)s;
    if (n < NDIM) {
        float xv = xh[node * (NDIM + HD) + n] * mask;
        x0[node * 3 + n] = xv;
        xcur[node * 3 + n] = xv;
    }
}

__global__ void radial_kernel(const float* __restrict__ x, float* __restrict__ out)
{
    int e = blockIdx.x * blockDim.x + threadIdx.x;
    if (e >= NE) return;
    int b = e >> 12, rem = e & 4095, i = rem >> 6, j = rem & 63;
    int r = b * 64 + i, c = b * 64 + j;
    float s = 0.f;
#pragma unroll
    for (int d = 0; d < 3; ++d) { float dd = x[r * 3 + d] - x[c * 3 + d]; s += dd * dd; }
    out[e] = s;
}

__global__ void edge_combine(const float* __restrict__ P, const float* __restrict__ Q,
                             const float* __restrict__ radial, const float* __restrict__ eattr,
                             const float* __restrict__ w_r, const float* __restrict__ w_e,
                             const float* __restrict__ b1, _Float16* __restrict__ m1)
{
    int e = blockIdx.x, n = threadIdx.x;   // NE x 256
    int b = e >> 12, rem = e & 4095, i = rem >> 6, j = rem & 63;
    int r = b * 64 + i, c = b * 64 + j;
    float v = P[(size_t)r * 256 + n] + Q[(size_t)c * 256 + n]
            + radial[e] * w_r[n] + eattr[e] * w_e[n] + b1[n];
    m1[(size_t)e * 256 + n] = (_Float16)silu_f(v);
}

__global__ void phi_kernel(const _Float16* __restrict__ m2, const float* __restrict__ cw2,
                           float* __restrict__ phi)
{
    int lane = threadIdx.x & 31;
    int e = blockIdx.x * (blockDim.x >> 5) + (threadIdx.x >> 5);
    const _Float16* p = m2 + (size_t)e * 256 + lane * 8;
    v8h v = *reinterpret_cast<const v8h*>(p);
    float s = 0.f;
#pragma unroll
    for (int i = 0; i < 8; ++i) s += (float)v[i] * cw2[lane * 8 + i];
#pragma unroll
    for (int off = 16; off > 0; off >>= 1) s += __shfl_xor(s, off, 32);
    if (lane == 0) phi[e] = s;
}

__global__ void agg_kernel(const _Float16* __restrict__ m, _Float16* __restrict__ hc)
{
    int node = blockIdx.x, n = threadIdx.x;   // 2048 x 256
    float s = 0.f;
    for (int j = 0; j < NN; ++j)
        s += (float)m[((size_t)node * NN + j) * 256 + n];
    hc[(size_t)node * 512 + 256 + n] = (_Float16)s;
}

__global__ void xupdate_kernel(const float* __restrict__ xin, const float* __restrict__ radial,
                               const float* __restrict__ phi, const float* __restrict__ em,
                               const float* __restrict__ nm, float* __restrict__ xout)
{
    int node = blockIdx.x * blockDim.x + threadIdx.x;
    if (node >= BN) return;
    int b = node >> 6;
    float xi0 = xin[node * 3], xi1 = xin[node * 3 + 1], xi2 = xin[node * 3 + 2];
    float a0 = 0.f, a1 = 0.f, a2 = 0.f;
    for (int j = 0; j < NN; ++j) {
        int e = node * NN + j;
        int c = b * 64 + j;
        float d0 = xi0 - xin[c * 3], d1 = xi1 - xin[c * 3 + 1], d2 = xi2 - xin[c * 3 + 2];
        float inv = 1.f / (sqrtf(radial[e] + 1e-8f) + 1.f);
        float w = phi[e] * em[e] * inv;
        a0 += d0 * w; a1 += d1 * w; a2 += d2 * w;
    }
    float mask = nm[node];
    xout[node * 3]     = (xi0 + a0) * mask;
    xout[node * 3 + 1] = (xi1 + a1) * mask;
    xout[node * 3 + 2] = (xi2 + a2) * mask;
}

__global__ void hout_kernel(const float* __restrict__ h, const float* __restrict__ W_out,
                            const float* __restrict__ b_out, const float* __restrict__ nm,
                            float* __restrict__ out)
{
    int node = blockIdx.x, n = threadIdx.x;   // 2048 x 32
    if (n >= HD) return;                      // drop last (9th) column
    float s = b_out[n];
    for (int k = 0; k < 256; ++k)
        s += h[(size_t)node * 256 + k] * W_out[k * (HD + 1) + n];
    out[(size_t)node * (NDIM + HD) + NDIM + n] = s * nm[node];
}

__global__ void vel_kernel(const float* __restrict__ xc, const float* __restrict__ x0,
                           const float* __restrict__ nm, float* __restrict__ out)
{
    __shared__ float sv[3][64];
    __shared__ float sc[64];
    int b = blockIdx.x, i = threadIdx.x;      // 32 x 64
    int node = b * 64 + i;
    float mask = nm[node];
    float v[3];
#pragma unroll
    for (int d = 0; d < 3; ++d) {
        v[d] = (xc[node * 3 + d] - x0[node * 3 + d]) * mask;
        sv[d][i] = v[d];
    }
    sc[i] = mask;
    __syncthreads();
    for (int off = 32; off > 0; off >>= 1) {
        if (i < off) {
#pragma unroll
            for (int d = 0; d < 3; ++d) sv[d][i] += sv[d][i + off];
            sc[i] += sc[i + off];
        }
        __syncthreads();
    }
    float inv = 1.f / sc[0];
#pragma unroll
    for (int d = 0; d < 3; ++d)
        out[(size_t)node * (NDIM + HD) + d] = v[d] - sv[d][0] * inv * mask;
}

// ---------------- host launcher ----------------
extern "C" void kernel_launch(void* const* d_in, const int* in_sizes, int n_in,
                              void* d_out, int out_size, void* d_ws, size_t ws_size,
                              hipStream_t stream)
{
    const float* t_s      = (const float*)d_in[0];
    const float* xh       = (const float*)d_in[1];
    const float* nm       = (const float*)d_in[2];
    const float* em       = (const float*)d_in[3];
    const float* W_emb    = (const float*)d_in[4];
    const float* b_emb    = (const float*)d_in[5];
    const float* W_out    = (const float*)d_in[6];
    const float* b_out    = (const float*)d_in[7];
    const float* edge_w1  = (const float*)d_in[8];
    const float* edge_b1  = (const float*)d_in[9];
    const float* edge_w2  = (const float*)d_in[10];
    const float* edge_b2  = (const float*)d_in[11];
    const float* node_w1  = (const float*)d_in[12];
    const float* node_b1  = (const float*)d_in[13];
    const float* node_w2  = (const float*)d_in[14];
    const float* node_b2  = (const float*)d_in[15];
    const float* coord_w1 = (const float*)d_in[16];
    const float* coord_b1 = (const float*)d_in[17];
    const float* coord_w2 = (const float*)d_in[18];
    float* out = (float*)d_out;

    // workspace layout
    char* ws = (char*)d_ws;
    size_t off = 0;
    auto take = [&](size_t bytes) { size_t o = off; off += (bytes + 255) & ~(size_t)255; return o; };
    float*    h     = (float*)   (ws + take((size_t)BN * 256 * 4));
    _Float16* hc    = (_Float16*)(ws + take((size_t)BN * 512 * 2));  // [h | agg] f16
    float*    Pbuf  = (float*)   (ws + take((size_t)BN * 256 * 4));
    float*    Qbuf  = (float*)   (ws + take((size_t)BN * 256 * 4));
    _Float16* n1buf = (_Float16*)(ws + take((size_t)BN * 256 * 2));
    _Float16* m1buf = (_Float16*)(ws + take((size_t)NE * 256 * 2)); // m1, reused as m2
    _Float16* Mbuf  = (_Float16*)(ws + take((size_t)NE * 256 * 2)); // m
    float*    rad   = (float*)   (ws + take((size_t)NE * 4));
    float*    eatt  = (float*)   (ws + take((size_t)NE * 4));
    float*    phi   = (float*)   (ws + take((size_t)NE * 4));
    float*    x0b   = (float*)   (ws + take((size_t)BN * 3 * 4));
    float*    xA    = (float*)   (ws + take((size_t)BN * 3 * 4));
    float*    xB    = (float*)   (ws + take((size_t)BN * 3 * 4));

    // per-layer transposed f16 weights: 0=ew1a 1=ew1b 2=ew2 3=cw1 4=nw1(K=512) 5=nw2
    _Float16* WT[NL][6];
    static const int WK[6] = {256, 256, 256, 256, 512, 256};
    for (int l = 0; l < NL; ++l)
        for (int w = 0; w < 6; ++w)
            WT[l][w] = (_Float16*)(ws + take((size_t)256 * WK[w] * 2));

    // convert + transpose all weights (one-time preamble)
    for (int l = 0; l < NL; ++l) {
        const float* ew1 = edge_w1 + (size_t)l * 514 * 256;
        wconv_kernel<<<256, 256, 0, stream>>>(ew1,                 WT[l][0], 256);
        wconv_kernel<<<256, 256, 0, stream>>>(ew1 + 256 * 256,     WT[l][1], 256);
        wconv_kernel<<<256, 256, 0, stream>>>(edge_w2 + (size_t)l * 65536, WT[l][2], 256);
        wconv_kernel<<<256, 256, 0, stream>>>(coord_w1 + (size_t)l * 65536, WT[l][3], 256);
        wconv_kernel<<<512, 256, 0, stream>>>(node_w1 + (size_t)l * 512 * 256, WT[l][4], 512);
        wconv_kernel<<<256, 256, 0, stream>>>(node_w2 + (size_t)l * 65536, WT[l][5], 256);
    }

    preprocess_embed<<<BN, 256, 0, stream>>>(xh, nm, t_s, W_emb, b_emb, h, hc, x0b, xA);
    radial_kernel<<<NE / 256, 256, 0, stream>>>(x0b, eatt);   // edge_attr from x0

    float* xcur = xA;
    float* xnxt = xB;

    const dim3 gN(BN / 64, 4);   // node-level GEMMs (M=2048)
    const dim3 gE(NE / 64, 4);   // edge-level GEMMs (M=131072)

    for (int l = 0; l < NL; ++l) {
        const float* ew1 = edge_w1 + (size_t)l * 514 * 256;

        radial_kernel<<<NE / 256, 256, 0, stream>>>(xcur, rad);

        // P = h @ W1a ; Q = h @ W1b   (h_f16 in hc cols 0..255, lda=512)
        gemm_wmma<0><<<gN, 128, 0, stream>>>(hc, 512, WT[l][0], 256,
                                             nullptr, nullptr, nullptr, Pbuf, nullptr, 0);
        gemm_wmma<0><<<gN, 128, 0, stream>>>(hc, 512, WT[l][1], 256,
                                             nullptr, nullptr, nullptr, Qbuf, nullptr, 0);

        // m1 = silu(P[row]+Q[col]+radial*w_r+eattr*w_e+b1)
        edge_combine<<<NE, 256, 0, stream>>>(Pbuf, Qbuf, rad, eatt,
                                             ew1 + 512 * 256, ew1 + 513 * 256,
                                             edge_b1 + l * 256, m1buf);

        // m = silu(m1 @ edge_w2 + b2) * em
        gemm_wmma<2><<<gE, 128, 0, stream>>>(m1buf, 256, WT[l][2], 256,
                                             edge_b2 + l * 256, em, nullptr,
                                             nullptr, Mbuf, 256);

        // m2 = silu(m @ coord_w1 + cb1)   (overwrites m1buf)
        gemm_wmma<1><<<gE, 128, 0, stream>>>(Mbuf, 256, WT[l][3], 256,
                                             coord_b1 + l * 256, nullptr, nullptr,
                                             nullptr, m1buf, 256);

        // phi = m2 @ coord_w2
        phi_kernel<<<NE / 8, 256, 0, stream>>>(m1buf, coord_w2 + l * 256, phi);

        // x update (ping-pong)
        xupdate_kernel<<<BN / 256, 256, 0, stream>>>(xcur, rad, phi, em, nm, xnxt);

        // agg into hc[:, 256:512]
        agg_kernel<<<BN, 256, 0, stream>>>(Mbuf, hc);

        // n1 = silu([h,agg] @ node_w1 + nb1)   (K=512)
        gemm_wmma<1><<<gN, 128, 0, stream>>>(hc, 512, WT[l][4], 512,
                                             node_b1 + l * 256, nullptr, nullptr,
                                             nullptr, n1buf, 256);

        // h = (h + n1 @ node_w2 + nb2) * nm  -> h (f32) and hc[:, 0:256] (f16)
        gemm_wmma<3><<<gN, 128, 0, stream>>>(n1buf, 256, WT[l][5], 256,
                                             node_b2 + l * 256, nm, h,
                                             h, hc, 512);

        float* tmp = xcur; xcur = xnxt; xnxt = tmp;
    }

    hout_kernel<<<BN, 32, 0, stream>>>(h, W_out, b_out, nm, out);
    vel_kernel<<<BSZ, 64, 0, stream>>>(xcur, x0b, nm, out);
}
